// My_model_v3_graclus_JK_4277787427385
// MI455X (gfx1250) — matrix operand order repaired and checked by reading, not compiled, not run on hardware
//
#include <hip/hip_runtime.h>
#include <hip/hip_bf16.h>

// ---------------------------------------------------------------------------
// Model constants (from the reference)
// ---------------------------------------------------------------------------
#define BATCH      256
#define N_ATOM     40
#define N_DRUG     (BATCH * N_ATOM)     // 10240
#define DIM_DRUG   128
#define DIM_CELL   64
#define D_ATT      256
#define FC_CELL    1024
#define OUT_CELL   256
#define FLAT       (89 * DIM_CELL)      // 5696
#define D_CAT      (OUT_CELL + 128)     // 384

typedef __attribute__((ext_vector_type(16))) __bf16 v16bf;
typedef __attribute__((ext_vector_type(8)))  __bf16 v8bf;
typedef __attribute__((ext_vector_type(8)))  float  v8f;

// LDS tile stride (elements). 40 elems = 80 B rows: 16B-aligned fragment reads,
// bank starts (20*r mod 64) = {0,4,...,60} across 16 rows -> conflict-free b128.
#define TSTR 40

// ---------------------------------------------------------------------------
// WMMA GEMM: C = [relu]( A(MxK,f32) @ W(KxN,f32) + bias [+ C] )
// Requirements (all call sites satisfy them): M % 128 == 0, N % 64 == 0.
// Block = 256 threads (8 waves) computing a 128x64 output macro-tile.
// Per K-step of 32: stage A(128x32) row-major and B(32x64) col-major as bf16
// in LDS; each wave owns a 2x2 grid of 16x16 subtiles (2 A frags x 2 B frags)
// and issues 4 WMMAs -> 32 WMMAs per block per K-step.
// ---------------------------------------------------------------------------
__global__ __launch_bounds__(256)
void k_wmma_gemm(const float* __restrict__ A, int lda,
                 const float* __restrict__ Bw, int ldb,
                 const float* __restrict__ bias,
                 float* __restrict__ C, int ldc,
                 int M, int N, int K, int relu, int addC)
{
    __shared__ __attribute__((aligned(16))) __bf16 As[128 * TSTR]; // As[r][k]
    __shared__ __attribute__((aligned(16))) __bf16 Bs[64 * TSTR];  // Bs[n][k] (transposed)

    const int t     = threadIdx.x;
    const int wave  = t >> 5;
    const int lane  = t & 31;
    const int lrow  = lane & 15;
    const int khalf = lane >> 4;

    const int nb   = N >> 6;
    const int bm   = blockIdx.x / nb;
    const int bn   = blockIdx.x % nb;
    const int row0 = bm << 7;                   // 128-row macro-tile
    const int col0 = bn << 6;                   // 64-col macro-tile

    // wave w -> 2x2 subtile block: rows {smA0, smA0+1}, cols {snB0, snB0+1}
    const int smA0 = (wave >> 1) << 1;          // 0,2,4,6
    const int snB0 = (wave & 1) << 1;           // 0,2

    const int arow0 = smA0 * 16 + lrow;         // A rows within macro-tile
    const int arow1 = arow0 + 16;
    const int nl0   = snB0 * 16 + lrow;         // B/C cols within macro-tile
    const int nl1   = nl0 + 16;
    const int ng0   = col0 + nl0;               // global columns
    const int ng1   = col0 + nl1;

    // staging coordinates
    const int sArow = t >> 1;                   // 0..127
    const int sAc16 = (t & 1) << 4;             // 0,16
    const int sBk   = t >> 3;                   // 0..31
    const int sBn8  = (t & 7) << 3;             // 0..56

    v8f acc00, acc01, acc10, acc11;
    {
        const float b0 = bias ? bias[ng0] : 0.0f;
        const float b1 = bias ? bias[ng1] : 0.0f;
        #pragma unroll
        for (int v = 0; v < 8; ++v) {
            const int mm0 = row0 + smA0 * 16 + v + (khalf << 3);
            const int mm1 = mm0 + 16;
            float i00 = b0, i01 = b1, i10 = b0, i11 = b1;
            if (addC) {
                i00 += C[(long)mm0 * ldc + ng0];
                i01 += C[(long)mm0 * ldc + ng1];
                i10 += C[(long)mm1 * ldc + ng0];
                i11 += C[(long)mm1 * ldc + ng1];
            }
            acc00[v] = i00; acc01[v] = i01; acc10[v] = i10; acc11[v] = i11;
        }
    }

    for (int k0 = 0; k0 < K; k0 += 32) {
        // ---- stage A tile (128x32) ----
        {
            __bf16* dst = &As[sArow * TSTR + sAc16];
            if (k0 + 32 <= K) {                      // block-uniform fast path
                const float* ap = A + (long)(row0 + sArow) * lda + k0 + sAc16;
                #pragma unroll
                for (int j = 0; j < 16; ++j) dst[j] = (__bf16)ap[j];
            } else {
                const float* abase = A + (long)(row0 + sArow) * lda;
                #pragma unroll
                for (int j = 0; j < 16; ++j) {
                    const int k = k0 + sAc16 + j;
                    const int kc = k < K ? k : (K - 1);   // clamp addr, mask value
                    float v = abase[kc];
                    dst[j] = (__bf16)(k < K ? v : 0.0f);
                }
            }
        }
        // ---- stage B tile (32x64), transposed into Bs[n][k] ----
        {
            if (k0 + 32 <= K) {
                const float* bp = Bw + (long)(k0 + sBk) * ldb + col0 + sBn8;
                #pragma unroll
                for (int j = 0; j < 8; ++j)
                    Bs[(sBn8 + j) * TSTR + sBk] = (__bf16)bp[j];
            } else {
                const int k = k0 + sBk;
                const int kc = k < K ? k : (K - 1);
                const float* bp = Bw + (long)kc * ldb + col0 + sBn8;
                #pragma unroll
                for (int j = 0; j < 8; ++j) {
                    float v = bp[j];
                    Bs[(sBn8 + j) * TSTR + sBk] = (__bf16)(k < K ? v : 0.0f);
                }
            }
        }
        __syncthreads();

        // ---- fragments from LDS (aligned 16B ds_load_b128s) ----
        union { v16bf v; v8bf h[2]; } af0, af1, bf0, bf1;
        af0.h[0] = *(const v8bf*)&As[arow0 * TSTR + (khalf << 3)];
        af0.h[1] = *(const v8bf*)&As[arow0 * TSTR + 16 + (khalf << 3)];
        af1.h[0] = *(const v8bf*)&As[arow1 * TSTR + (khalf << 3)];
        af1.h[1] = *(const v8bf*)&As[arow1 * TSTR + 16 + (khalf << 3)];
        bf0.h[0] = *(const v8bf*)&Bs[nl0 * TSTR + (khalf << 3)];
        bf0.h[1] = *(const v8bf*)&Bs[nl0 * TSTR + 16 + (khalf << 3)];
        bf1.h[0] = *(const v8bf*)&Bs[nl1 * TSTR + (khalf << 3)];
        bf1.h[1] = *(const v8bf*)&Bs[nl1 * TSTR + 16 + (khalf << 3)];

        acc00 = __builtin_amdgcn_wmma_f32_16x16x32_bf16(
            false, af0.v, false, bf0.v, (short)0, acc00, false, false);
        acc01 = __builtin_amdgcn_wmma_f32_16x16x32_bf16(
            false, af0.v, false, bf1.v, (short)0, acc01, false, false);
        acc10 = __builtin_amdgcn_wmma_f32_16x16x32_bf16(
            false, af1.v, false, bf0.v, (short)0, acc10, false, false);
        acc11 = __builtin_amdgcn_wmma_f32_16x16x32_bf16(
            false, af1.v, false, bf1.v, (short)0, acc11, false, false);

        __syncthreads();
    }

    #pragma unroll
    for (int v = 0; v < 8; ++v) {
        const int mm0 = row0 + smA0 * 16 + v + (khalf << 3);
        const int mm1 = mm0 + 16;
        float r00 = acc00[v], r01 = acc01[v], r10 = acc10[v], r11 = acc11[v];
        if (relu) {
            r00 = fmaxf(r00, 0.0f); r01 = fmaxf(r01, 0.0f);
            r10 = fmaxf(r10, 0.0f); r11 = fmaxf(r11, 0.0f);
        }
        C[(long)mm0 * ldc + ng0] = r00;
        C[(long)mm0 * ldc + ng1] = r01;
        C[(long)mm1 * ldc + ng0] = r10;
        C[(long)mm1 * ldc + ng1] = r11;
    }
}

// ---------------------------------------------------------------------------
// Elementwise / graph kernels
// ---------------------------------------------------------------------------
__global__ void k_fill(float* p, long n, float v) {
    long i = (long)blockIdx.x * blockDim.x + threadIdx.x;
    if (i < n) p[i] = v;
}

__global__ void k_copy(const float* __restrict__ s, float* __restrict__ d, long n) {
    long i = (long)blockIdx.x * blockDim.x + threadIdx.x;
    if (i < n) d[i] = s[i];
}

// h[dst] += x[src] over edge list ei = [src row | dst row] (shape 2 x E flat)
__global__ void k_edge_add(const float* __restrict__ x, const int* __restrict__ ei,
                           int E, int D, float* __restrict__ h) {
    long i = (long)blockIdx.x * blockDim.x + threadIdx.x;
    if (i >= (long)E * D) return;
    const int e = (int)(i / D);
    const int d = (int)(i % D);
    const int src = ei[e];
    const int dst = ei[E + e];
    atomicAdd(&h[(long)dst * D + d], x[(long)src * D + d]);
}

__global__ void k_edge_cnt(const int* __restrict__ ei, int E, float* __restrict__ cnt) {
    int e = blockIdx.x * blockDim.x + threadIdx.x;
    if (e < E) atomicAdd(&cnt[ei[E + e]], 1.0f);
}

__global__ void k_div_cnt(float* __restrict__ s, const float* __restrict__ cnt, long n, int D) {
    long i = (long)blockIdx.x * blockDim.x + threadIdx.x;
    if (i >= n * (long)D) return;
    const int r = (int)(i / D);
    s[i] /= fmaxf(cnt[r], 1.0f);
}

// segment_max via int-bitcast atomicMax (inputs are post-ReLU, >= 0; out pre-filled 0)
__global__ void k_cluster_max(const float* __restrict__ x, const int* __restrict__ cl,
                              long nIn, int D, float* __restrict__ out) {
    long i = (long)blockIdx.x * blockDim.x + threadIdx.x;
    if (i >= nIn * (long)D) return;
    const int r = (int)(i / D);
    const int d = (int)(i % D);
    atomicMax((int*)&out[(long)cl[r] * D + d], __float_as_int(x[i]));
}

// Per-column batchnorm stats (population var, matches jnp.var ddof=0)
__global__ void k_bn_stats(const float* __restrict__ x, int M, int D,
                           float* __restrict__ mean, float* __restrict__ var) {
    const int d = blockIdx.x;
    const int t = threadIdx.x;
    __shared__ float ss[256];
    __shared__ float sq[256];
    float s = 0.0f, q = 0.0f;
    for (int r = t; r < M; r += 256) {
        const float v = x[(long)r * D + d];
        s += v; q += v * v;
    }
    ss[t] = s; sq[t] = q; __syncthreads();
    for (int k = 128; k > 0; k >>= 1) {
        if (t < k) { ss[t] += ss[t + k]; sq[t] += sq[t + k]; }
        __syncthreads();
    }
    if (t == 0) {
        const float m = ss[0] / (float)M;
        mean[d] = m;
        var[d]  = sq[0] / (float)M - m * m;
    }
}

__global__ void k_bn_apply(float* __restrict__ x, long M, int D,
                           const float* __restrict__ mean, const float* __restrict__ var,
                           const float* __restrict__ g, const float* __restrict__ b) {
    long i = (long)blockIdx.x * blockDim.x + threadIdx.x;
    if (i >= M * (long)D) return;
    const int d = (int)(i % D);
    x[i] = (x[i] - mean[d]) * rsqrtf(var[d] + 1e-5f) * g[d] + b[d];
}

// Drug graph pooling: drug_batch is contiguous (40 atoms per graph) -> direct loop
__global__ void k_drug_pool(const float* __restrict__ x, float* __restrict__ kb, int layer) {
    int i = blockIdx.x * blockDim.x + threadIdx.x;
    if (i >= BATCH * DIM_DRUG) return;
    const int b = i / DIM_DRUG;
    const int d = i % DIM_DRUG;
    const float* base = x + (long)b * N_ATOM * DIM_DRUG + d;
    float s = 0.0f, m = -3.4e38f;
    for (int a = 0; a < N_ATOM; ++a) {
        const float v = base[(long)a * DIM_DRUG];
        s += v; m = fmaxf(m, v);
    }
    kb[((long)(b * 3 + layer)) * D_ATT + d]            = s / (float)N_ATOM; // mean half
    kb[((long)(b * 3 + layer)) * D_ATT + DIM_DRUG + d] = m;                 // max half
}

// Attention over 3 JK levels: scores = q.K_l / 16, softmax, weighted sum of K
__global__ void k_attention(const float* __restrict__ q, const float* __restrict__ kb,
                            float* __restrict__ out) {
    const int b = blockIdx.x;
    const int d = threadIdx.x;            // 256 threads
    __shared__ float red[256];
    __shared__ float aw[3];
    const float qd = q[(long)b * D_ATT + d];
    for (int l = 0; l < 3; ++l) {
        red[d] = qd * kb[((long)(b * 3 + l)) * D_ATT + d];
        __syncthreads();
        for (int s = 128; s > 0; s >>= 1) {
            if (d < s) red[d] += red[d + s];
            __syncthreads();
        }
        if (d == 0) aw[l] = red[0] * (1.0f / 16.0f);  // / sqrt(256)
        __syncthreads();
    }
    if (d == 0) {
        const float mx = fmaxf(aw[0], fmaxf(aw[1], aw[2]));
        const float e0 = expf(aw[0] - mx), e1 = expf(aw[1] - mx), e2 = expf(aw[2] - mx);
        const float inv = 1.0f / (e0 + e1 + e2);
        aw[0] = e0 * inv; aw[1] = e1 * inv; aw[2] = e2 * inv;
    }
    __syncthreads();
    out[(long)b * D_ATT + d] =
        aw[0] * kb[((long)(b * 3 + 0)) * D_ATT + d] +
        aw[1] * kb[((long)(b * 3 + 1)) * D_ATT + d] +
        aw[2] * kb[((long)(b * 3 + 2)) * D_ATT + d];
}

// Final 384 -> 1 projection
__global__ void k_dot_out(const float* __restrict__ h, const float* __restrict__ w,
                          const float* __restrict__ bias, float* __restrict__ out) {
    const int b = blockIdx.x;
    const int t = threadIdx.x;            // 128 threads
    __shared__ float red[128];
    float s = 0.0f;
    for (int i = t; i < D_CAT; i += 128) s += h[(long)b * D_CAT + i] * w[i];
    red[t] = s; __syncthreads();
    for (int k = 64; k > 0; k >>= 1) {
        if (t < k) red[t] += red[t + k];
        __syncthreads();
    }
    if (t == 0) out[b] = red[0] + bias[0];
}

// ---------------------------------------------------------------------------
// Host-side launch
// ---------------------------------------------------------------------------
static inline unsigned cdiv(long n, int b) { return (unsigned)((n + b - 1) / b); }

static void launch_gemm(const float* A, int lda, const float* W, int ldb, const float* bias,
                        float* C, int ldc, int M, int N, int K, int relu, int addC,
                        hipStream_t st) {
    // M % 128 == 0 and N % 64 == 0 hold for every call site.
    const unsigned blocks = (unsigned)((M / 128) * (N / 64));
    k_wmma_gemm<<<dim3(blocks), 256, 0, st>>>(A, lda, W, ldb, bias, C, ldc,
                                              M, N, K, relu, addC);
}

extern "C" void kernel_launch(void* const* d_in, const int* in_sizes, int n_in,
                              void* d_out, int out_size, void* d_ws, size_t ws_size,
                              hipStream_t stream) {
    // ---- input mapping (setup_inputs dict order; params dict flattened) ----
    const float* drug_x = (const float*)d_in[0];   // (10240, 77)
    const float* cell_x = (const float*)d_in[1];   // (180736, 3)
    auto P = [&](int i) { return (const float*)d_in[i]; };
    // params: 2 + i*6 : gin{i} W1,b1,W2,b2,g,b   (i = 0..2)
    //         20 + i*5: sage{i} Wl,bl,Wr,g,b     (i = 0..2)
    //         35: att_Wq  36: att_Wo  37: att_bo
    //         38: ce_W1   39: ce_b1   40: ce_W2   41: ce_b2
    //         42..47: rg_W1,rg_b1,rg_W2,rg_b2,rg_W3,rg_b3
    const int* ei_drug  = (const int*)d_in[48];
    const int* cell_ei[3] = { (const int*)d_in[50], (const int*)d_in[51], (const int*)d_in[52] };
    const int* clus[3]    = { (const int*)d_in[53], (const int*)d_in[54], (const int*)d_in[55] };
    const int Edrug = in_sizes[48] / 2;
    const int Ecell[3] = { in_sizes[50] / 2, in_sizes[51] / 2, in_sizes[52] / 2 };

    // ---- workspace carve-up ----
    float* w = (float*)d_ws;
    float* H   = w; w += (long)N_DRUG * DIM_DRUG;
    float* T   = w; w += (long)N_DRUG * DIM_DRUG;
    float* Y0  = w; w += (long)N_DRUG * DIM_DRUG;
    float* Y1  = w; w += (long)N_DRUG * DIM_DRUG;
    float* KB  = w; w += (long)BATCH * 3 * D_ATT;
    float* CX  = w; w += (long)BATCH * 706 * DIM_CELL;
    float* CS  = w; w += (long)BATCH * 706 * DIM_CELL;
    float* CO  = w; w += (long)BATCH * 706 * DIM_CELL;
    float* CNT = w; w += (long)BATCH * 706;
    float* MEAN= w; w += 128;
    float* VAR = w; w += 128;
    float* Q   = w; w += (long)BATCH * D_ATT;
    float* XD  = w; w += (long)BATCH * D_ATT;
    float* H1  = w; w += (long)BATCH * FC_CELL;
    float* HC  = w; w += (long)BATCH * D_CAT;
    float* R1  = w; w += (long)BATCH * D_CAT;
    float* R2  = w; w += (long)BATCH * D_CAT;
    (void)ws_size; (void)n_in; (void)out_size;

    // ================= Drug branch: 3 x GIN + BN + JK pooling =================
    {
        const float* x = drug_x;
        int D = 77;
        float* Youts[3] = { Y0, Y1, Y0 };
        for (int i = 0; i < 3; ++i) {
            const float* W1 = P(2 + i * 6 + 0);
            const float* b1 = P(2 + i * 6 + 1);
            const float* W2 = P(2 + i * 6 + 2);
            const float* b2 = P(2 + i * 6 + 3);
            const float* g  = P(2 + i * 6 + 4);
            const float* bb = P(2 + i * 6 + 5);
            float* Y = Youts[i];
            // h = x + segment_sum(x[src] -> dst)
            k_copy<<<cdiv((long)N_DRUG * D, 256), 256, 0, stream>>>(x, H, (long)N_DRUG * D);
            k_edge_add<<<cdiv((long)Edrug * D, 256), 256, 0, stream>>>(x, ei_drug, Edrug, D, H);
            // t = relu(h @ W1 + b1) ; y = relu(t @ W2 + b2)
            launch_gemm(H, D, W1, DIM_DRUG, b1, T, DIM_DRUG, N_DRUG, DIM_DRUG, D, 1, 0, stream);
            launch_gemm(T, DIM_DRUG, W2, DIM_DRUG, b2, Y, DIM_DRUG, N_DRUG, DIM_DRUG, DIM_DRUG, 1, 0, stream);
            // batchnorm
            k_bn_stats<<<DIM_DRUG, 256, 0, stream>>>(Y, N_DRUG, DIM_DRUG, MEAN, VAR);
            k_bn_apply<<<cdiv((long)N_DRUG * DIM_DRUG, 256), 256, 0, stream>>>(Y, N_DRUG, DIM_DRUG, MEAN, VAR, g, bb);
            // JK pooling (mean | max) -> KB[:, i, :]
            k_drug_pool<<<cdiv(BATCH * DIM_DRUG, 256), 256, 0, stream>>>(Y, KB, i);
            x = Y; D = DIM_DRUG;
        }
    }

    // ================= Cell branch: 3 x SAGE + graclus max-pool + BN ==========
    {
        const int NIN[3]  = { BATCH * 706, BATCH * 353, BATCH * 177 };
        const int NOUT[3] = { BATCH * 353, BATCH * 177, BATCH * 89  };
        const float* x = cell_x;
        int D = 3;
        for (int i = 0; i < 3; ++i) {
            const float* Wl = P(20 + i * 5 + 0);
            const float* bl = P(20 + i * 5 + 1);
            const float* Wr = P(20 + i * 5 + 2);
            const float* g  = P(20 + i * 5 + 3);
            const float* bb = P(20 + i * 5 + 4);
            // mean aggregation
            k_fill<<<cdiv((long)NIN[i] * D, 256), 256, 0, stream>>>(CS, (long)NIN[i] * D, 0.0f);
            k_fill<<<cdiv((long)NIN[i], 256), 256, 0, stream>>>(CNT, (long)NIN[i], 0.0f);
            k_edge_add<<<cdiv((long)Ecell[i] * D, 256), 256, 0, stream>>>(x, cell_ei[i], Ecell[i], D, CS);
            k_edge_cnt<<<cdiv((long)Ecell[i], 256), 256, 0, stream>>>(cell_ei[i], Ecell[i], CNT);
            k_div_cnt<<<cdiv((long)NIN[i] * D, 256), 256, 0, stream>>>(CS, CNT, (long)NIN[i], D);
            // out = relu(agg @ Wl + bl + x @ Wr)
            launch_gemm(CS, D, Wl, DIM_CELL, bl, CO, DIM_CELL, NIN[i], DIM_CELL, D, 0, 0, stream);
            launch_gemm(x,  D, Wr, DIM_CELL, nullptr, CO, DIM_CELL, NIN[i], DIM_CELL, D, 1, 1, stream);
            // graclus max pooling (post-ReLU values >= 0 -> int atomicMax valid)
            k_fill<<<cdiv((long)NOUT[i] * DIM_CELL, 256), 256, 0, stream>>>(CX, (long)NOUT[i] * DIM_CELL, 0.0f);
            k_cluster_max<<<cdiv((long)NIN[i] * DIM_CELL, 256), 256, 0, stream>>>(CO, clus[i], (long)NIN[i], DIM_CELL, CX);
            // batchnorm
            k_bn_stats<<<DIM_CELL, 256, 0, stream>>>(CX, NOUT[i], DIM_CELL, MEAN, VAR);
            k_bn_apply<<<cdiv((long)NOUT[i] * DIM_CELL, 256), 256, 0, stream>>>(CX, NOUT[i], DIM_CELL, MEAN, VAR, g, bb);
            x = CX; D = DIM_CELL;
        }
    }

    // ================= Head: attention + cell MLP + regressor =================
    // q = x_cell @ att_Wq
    launch_gemm(CX, FLAT, P(35), D_ATT, nullptr, Q, D_ATT, BATCH, D_ATT, FLAT, 0, 0, stream);
    // attention over 3 JK levels
    k_attention<<<BATCH, 256, 0, stream>>>(Q, KB, XD);
    // x_drug = (a.K) @ att_Wo + att_bo -> HC[:, 0:128]
    launch_gemm(XD, D_ATT, P(36), 128, P(37), HC, D_CAT, BATCH, 128, D_ATT, 0, 0, stream);
    // h_cell = relu(relu(x_cell @ ce_W1 + ce_b1) @ ce_W2 + ce_b2) -> HC[:, 128:384]
    launch_gemm(CX, FLAT, P(38), FC_CELL, P(39), H1, FC_CELL, BATCH, FC_CELL, FLAT, 1, 0, stream);
    launch_gemm(H1, FC_CELL, P(40), OUT_CELL, P(41), HC + 128, D_CAT, BATCH, OUT_CELL, FC_CELL, 1, 0, stream);
    // regressor
    launch_gemm(HC, D_CAT, P(42), D_CAT, P(43), R1, D_CAT, BATCH, D_CAT, D_CAT, 1, 0, stream);
    launch_gemm(R1, D_CAT, P(44), D_CAT, P(45), R2, D_CAT, BATCH, D_CAT, D_CAT, 1, 0, stream);
    k_dot_out<<<BATCH, 128, 0, stream>>>(R2, P(46), P(47), (float*)d_out);
}
